// Model_63213328663230
// MI455X (gfx1250) — matrix operand order repaired and checked by reading, not compiled, not run on hardware
//
#include <hip/hip_runtime.h>
#include <math.h>
#include <stdint.h>

// ---------------- model constants ----------------
#define BSZ   4
#define NTOK  2048
#define SPC   3
#define CDIM  32
#define OUTC  4
#define POCT  5
#define DDIM  223
#define DP    224          // DDIM padded to 32
#define NBLK  4
#define NH    8
#define NEXP  4
#define ENCS  33
#define NDIM  256
#define HD    32
#define FFN   512
#define DTs   0.05f
#define MTOK  (BSZ*NTOK)   // 8192
#define FEATW 925
#define FEATP 928

typedef __attribute__((ext_vector_type(16))) _Float16 v16h;
typedef __attribute__((ext_vector_type(8)))  _Float16 v8h;
typedef __attribute__((ext_vector_type(8)))  float    v8f;
typedef __attribute__((ext_vector_type(4)))  int      v4i;

enum { ACT_NONE = 0, ACT_SILU = 1, ACT_SIGMOID = 2 };

// ---- CDNA5 async global->LDS copy (guarded; fallback: direct LDS store) ----
#if defined(__has_builtin)
#  if __has_builtin(__builtin_amdgcn_global_load_async_to_lds_b128)
#    define HAVE_ASYNC_LDS 1
#  endif
#endif
#ifndef HAVE_ASYNC_LDS
#  define HAVE_ASYNC_LDS 0
#endif

#if HAVE_ASYNC_LDS
typedef __attribute__((address_space(1))) v4i v4i_g;
typedef __attribute__((address_space(3))) v4i v4i_l;
#endif

__device__ __forceinline__ void async_cp16(const float* g, float* l) {
#if HAVE_ASYNC_LDS
  __builtin_amdgcn_global_load_async_to_lds_b128(
      (v4i_g*)(uintptr_t)g, (v4i_l*)(uintptr_t)l, 0, 0);
#else
  *(float4*)l = *(const float4*)g;
#endif
}

__device__ __forceinline__ void wait_async0() {
#if HAVE_ASYNC_LDS
#  if __has_builtin(__builtin_amdgcn_s_wait_asynccnt)
  __builtin_amdgcn_s_wait_asynccnt(0);
#  else
  asm volatile("s_wait_asynccnt 0" ::: "memory");
#  endif
#else
  asm volatile("s_wait_dscnt 0" ::: "memory");
#endif
}

__device__ __forceinline__ v8f wmma16x16x32(v16h a, v16h b, v8f c) {
  return __builtin_amdgcn_wmma_f32_16x16x32_f16(false, a, false, b, (short)0, c,
                                                false, false);
}

// Convert two contiguous 8-float groups (p[0..7], p[16..23]) into an A fragment.
__device__ __forceinline__ v16h loadA_frag(const float* __restrict__ p) {
  const float4* q = (const float4*)p;
  float4 x0 = q[0], x1 = q[1], x2 = q[4], x3 = q[5];
  v16h a;
  a[0]=(_Float16)x0.x; a[1]=(_Float16)x0.y; a[2]=(_Float16)x0.z; a[3]=(_Float16)x0.w;
  a[4]=(_Float16)x1.x; a[5]=(_Float16)x1.y; a[6]=(_Float16)x1.z; a[7]=(_Float16)x1.w;
  a[8]=(_Float16)x2.x; a[9]=(_Float16)x2.y; a[10]=(_Float16)x2.z; a[11]=(_Float16)x2.w;
  a[12]=(_Float16)x3.x; a[13]=(_Float16)x3.y; a[14]=(_Float16)x3.z; a[15]=(_Float16)x3.w;
  return a;
}

// 16 contiguous f32 -> v16h. 16B aligned.
__device__ __forceinline__ v16h cvt16(const float* __restrict__ p) {
  const float4* q = (const float4*)p;
  float4 x0 = q[0], x1 = q[1], x2 = q[2], x3 = q[3];
  v16h a;
  a[0]=(_Float16)x0.x; a[1]=(_Float16)x0.y; a[2]=(_Float16)x0.z; a[3]=(_Float16)x0.w;
  a[4]=(_Float16)x1.x; a[5]=(_Float16)x1.y; a[6]=(_Float16)x1.z; a[7]=(_Float16)x1.w;
  a[8]=(_Float16)x2.x; a[9]=(_Float16)x2.y; a[10]=(_Float16)x2.z; a[11]=(_Float16)x2.w;
  a[12]=(_Float16)x3.x; a[13]=(_Float16)x3.y; a[14]=(_Float16)x3.z; a[15]=(_Float16)x3.w;
  return a;
}

// ---------------- weight pre-pack: f32 [K,N] row-major -> f16 WMMA-B order ----------------
__global__ void pack_w(const float* __restrict__ W, _Float16* __restrict__ WP,
                       int K, int N, int Np, int total) {
  int i = blockIdx.x * blockDim.x + threadIdx.x;
  if (i >= total) return;
  int pos = i & 31;
  int c = (i >> 5) % Np;
  int kt = i / (32 * Np);
  int k = kt * 32 + (pos & 7) + ((pos >> 4) & 1) * 8 + ((pos >> 3) & 1) * 16;
  WP[i] = (k < K && c < N) ? (_Float16)W[(size_t)k * N + c] : (_Float16)0.0f;
}

// ---------------- WMMA GEMM: C = epi(A[M,K] @ W[K,N] + bias), 32x32 tile / wave ----------------
__device__ __forceinline__ void store_tile(
    v8f acc, int row0, int col0, const float* __restrict__ bias,
    float* __restrict__ C, int ldc, int M, int N, int act,
    const float* __restrict__ rowscale, int rs_stride, int rs_off,
    int accumulate, float outscale, int lane) {
  int c = col0 + (lane & 15);
  int rbase = row0 + ((lane >> 4) << 3);
  if (c >= N) return;
  float bv = bias ? bias[c] : 0.0f;
#pragma unroll
  for (int i = 0; i < 8; ++i) {
    int r = rbase + i;
    if (r >= M) break;
    float v = acc[i] + bv;
    if (act == ACT_SILU)         v = v / (1.0f + __expf(-v));
    else if (act == ACT_SIGMOID) v = 1.0f / (1.0f + __expf(-v));
    if (rowscale) v *= rowscale[(size_t)r * rs_stride + rs_off];
    v *= outscale;
    size_t idx = (size_t)r * ldc + c;
    if (accumulate) v += C[idx];
    C[idx] = v;
  }
}

__global__ void __launch_bounds__(32)
gemm_wmma(const float* __restrict__ A, int lda,
          const _Float16* __restrict__ WP, int npad,
          const float* __restrict__ bias, float* __restrict__ C, int ldc,
          int M, int K, int N, int act,
          const float* __restrict__ rowscale, int rs_stride, int rs_off,
          int accumulate, float outscale) {
  int lane = threadIdx.x & 31;
  int col0 = blockIdx.x * 32;
  int row0 = blockIdx.y * 32;

  int kb = (lane & 16) ? 8 : 0;
  const float* a0p = A + (size_t)(row0 + (lane & 15)) * lda + kb;
  const float* a1p = a0p + (size_t)16 * lda;

  int cB = col0 + (lane & 15);
  const _Float16* b0p = WP + (size_t)cB * 32 + ((lane & 16) ? 16 : 0);
  const _Float16* b1p = b0p + 16 * 32;
  size_t bstep = (size_t)npad * 32;

  v8f acc00 = {}, acc01 = {}, acc10 = {}, acc11 = {};
  for (int k0 = 0; k0 < K; k0 += 32) {
    v16h a0 = loadA_frag(a0p + k0);
    v16h a1 = loadA_frag(a1p + k0);
    v16h b0 = *(const v16h*)b0p;
    v16h b1 = *(const v16h*)b1p;
    b0p += bstep; b1p += bstep;
    acc00 = wmma16x16x32(a0, b0, acc00);
    acc01 = wmma16x16x32(a0, b1, acc01);
    acc10 = wmma16x16x32(a1, b0, acc10);
    acc11 = wmma16x16x32(a1, b1, acc11);
  }
  store_tile(acc00, row0,      col0,      bias, C, ldc, M, N, act, rowscale, rs_stride, rs_off, accumulate, outscale, lane);
  store_tile(acc01, row0,      col0 + 16, bias, C, ldc, M, N, act, rowscale, rs_stride, rs_off, accumulate, outscale, lane);
  store_tile(acc10, row0 + 16, col0,      bias, C, ldc, M, N, act, rowscale, rs_stride, rs_off, accumulate, outscale, lane);
  store_tile(acc11, row0 + 16, col0 + 16, bias, C, ldc, M, N, act, rowscale, rs_stride, rs_off, accumulate, outscale, lane);
}

// ---------------- flash attention, one wave per (b, h, 16-row q tile) ----------------
__global__ void __launch_bounds__(32)
flash_attn(const float* __restrict__ Q, const float* __restrict__ Km,
           const float* __restrict__ Vm, float* __restrict__ O, int n) {
  __shared__ _Float16 Pst[16 * 32];
  __shared__ float    Vst[32 * 32];   // V tile staged via async global->LDS
  int qt = blockIdx.x, h = blockIdx.y, b = blockIdx.z;
  int lane = threadIdx.x & 31;
  const float* Qb = Q + (size_t)b * n * NDIM + h * HD;
  const float* Kb = Km + (size_t)b * n * NDIM + h * HD;
  const float* Vb = Vm + (size_t)b * n * NDIM + h * HD;

  const float scale = 0.17677669529663687f;  // HD^-0.5
  int qr = qt * 16 + (lane & 15);
  int kbq = (lane < 16) ? 0 : 8;
  v16h qa = loadA_frag(Qb + (size_t)qr * NDIM + kbq);
#pragma unroll
  for (int e = 0; e < 16; ++e) qa[e] = (_Float16)((float)qa[e] * scale);

  v8f o0 = {}, o1 = {};
  float m[8], l[8];
#pragma unroll
  for (int i = 0; i < 8; ++i) { m[i] = -1e30f; l[i] = 0.0f; }

  int colc = lane & 15;
  int rsel = (lane >> 4) << 3;
  int dstart = (lane < 16) ? 0 : 16;

  for (int kt = 0; kt < n; kt += 32) {
    // stage V tile (32 keys x 32 dims, f32): lane copies its key row in 8x16B chunks
    {
      const float* vrow = Vb + (size_t)(kt + lane) * NDIM;
      float* lrow = &Vst[lane * 32];
#pragma unroll
      for (int j = 0; j < 8; ++j) async_cp16(vrow + j * 4, lrow + j * 4);
    }

    // S = Q K^T
    int key0 = kt + colc;
    v16h kb0 = loadA_frag(Kb + (size_t)key0 * NDIM + kbq);
    v16h kb1 = loadA_frag(Kb + (size_t)(key0 + 16) * NDIM + kbq);
    v8f z = {};
    v8f s0 = wmma16x16x32(qa, kb0, z);
    v8f s1 = wmma16x16x32(qa, kb1, z);

    // online softmax per row
#pragma unroll
    for (int i = 0; i < 8; ++i) {
      float mx = fmaxf(s0[i], s1[i]);
      mx = fmaxf(mx, __shfl_xor(mx, 1, 32));
      mx = fmaxf(mx, __shfl_xor(mx, 2, 32));
      mx = fmaxf(mx, __shfl_xor(mx, 4, 32));
      mx = fmaxf(mx, __shfl_xor(mx, 8, 32));
      float mn = fmaxf(m[i], mx);
      float al = __expf(m[i] - mn);
      float p0 = __expf(s0[i] - mn);
      float p1 = __expf(s1[i] - mn);
      float rs = p0 + p1;
      rs += __shfl_xor(rs, 1, 32);
      rs += __shfl_xor(rs, 2, 32);
      rs += __shfl_xor(rs, 4, 32);
      rs += __shfl_xor(rs, 8, 32);
      m[i] = mn;
      l[i] = l[i] * al + rs;
      s0[i] = p0; s1[i] = p1;
      o0[i] *= al; o1[i] *= al;
    }

    // stage P (16x32 f16) and reload in A layout (two contiguous 16B runs per lane)
#pragma unroll
    for (int i = 0; i < 8; ++i) {
      Pst[(rsel + i) * 32 + colc]      = (_Float16)s0[i];
      Pst[(rsel + i) * 32 + 16 + colc] = (_Float16)s1[i];
    }
    asm volatile("s_wait_dscnt 0" ::: "memory");
    const _Float16* pr = &Pst[(lane & 15) * 32 + kbq];
    v8h plo = *(const v8h*)pr;
    v8h phi = *(const v8h*)(pr + 16);
    v16h pa;
#pragma unroll
    for (int e = 0; e < 8; ++e) { pa[e] = plo[e]; pa[8 + e] = phi[e]; }

    // V fragments from LDS (B layout: K = local key (e&7)+kbq+16*(e>>3), col = colc)
    wait_async0();
    v16h vb0, vb1;
#pragma unroll
    for (int e = 0; e < 16; ++e) {
      const float* Vr = &Vst[((e & 7) + kbq + ((e >> 3) << 4)) * 32];
      vb0[e] = (_Float16)Vr[colc];
      vb1[e] = (_Float16)Vr[16 + colc];
    }
    o0 = wmma16x16x32(pa, vb0, o0);
    o1 = wmma16x16x32(pa, vb1, o1);
  }

#pragma unroll
  for (int i = 0; i < 8; ++i) {
    float inv = 1.0f / l[i];
    int row = qt * 16 + rsel + i;
    float* Ob = O + (size_t)(b * n + row) * NDIM + h * HD;
    Ob[colc]      = o0[i] * inv;
    Ob[16 + colc] = o1[i] * inv;
  }
}

// ---------------- layernorm over 256 cols, one wave per row ----------------
__global__ void __launch_bounds__(256)
layernorm256(const float* __restrict__ X, const float* __restrict__ g,
             const float* __restrict__ bt, float* __restrict__ Y, int M) {
  int wave = threadIdx.x >> 5, lane = threadIdx.x & 31;
  int row = blockIdx.x * 8 + wave;
  if (row >= M) return;
  const float* x = X + (size_t)row * NDIM;
  float v[8], s = 0.0f;
#pragma unroll
  for (int i = 0; i < 8; ++i) { v[i] = x[lane + 32 * i]; s += v[i]; }
#pragma unroll
  for (int off = 1; off < 32; off <<= 1) s += __shfl_xor(s, off, 32);
  float mean = s * (1.0f / NDIM);
  float var = 0.0f;
#pragma unroll
  for (int i = 0; i < 8; ++i) { float d = v[i] - mean; var += d * d; }
#pragma unroll
  for (int off = 1; off < 32; off <<= 1) var += __shfl_xor(var, off, 32);
  float inv = rsqrtf(var * (1.0f / NDIM) + 1e-5f);
  float* y = Y + (size_t)row * NDIM;
#pragma unroll
  for (int i = 0; i < 8; ++i) {
    int c = lane + 32 * i;
    y[c] = (v[i] - mean) * inv * g[c] + bt[c];
  }
}

// ---------------- elementwise helpers ----------------
__global__ void fourier_g(const float* __restrict__ in, float* __restrict__ out,
                          int B, int Din, int ldo) {
  int b = blockIdx.x * blockDim.x + threadIdx.x;
  if (b >= B) return;
  float* o = out + (size_t)b * ldo;
  const float* p = in + (size_t)b * Din;
  for (int s = 0; s < Din; ++s) {
    float pv = p[s];
#pragma unroll
    for (int k = 0; k < POCT; ++k) {
      float a = pv * (float)(1 << k) * 3.14159265358979323846f;
      o[s * 10 + k]     = cosf(a);
      o[s * 10 + 5 + k] = sinf(a);
    }
  }
  for (int s = 0; s < Din; ++s) o[10 * Din + s] = p[s];
  for (int s = 11 * Din; s < ldo; ++s) o[s] = 0.0f;
}

__global__ void cat7_k(const float* __restrict__ st, const float* __restrict__ np,
                       float* __restrict__ out, int M) {
  int i = blockIdx.x * blockDim.x + threadIdx.x;
  if (i >= M * 32) return;
  int t = i >> 5, c = i & 31;
  float v = 0.0f;
  if (c < 4) v = st[t * 4 + c];
  else if (c < 7) v = np[t * 3 + (c - 4)];
  out[i] = v;
}

__global__ void addbc_k(float* __restrict__ V, const float* __restrict__ TV,
                        const float* __restrict__ CV, int M) {
  int i = blockIdx.x * blockDim.x + threadIdx.x;
  if (i >= M * DDIM) return;
  int t = i / DDIM, c = i - t * DDIM;
  int b = t >> 11;
  V[(size_t)t * DP + c] += TV[b * DP + c] + CV[b * DP + c];
}

__global__ void concat_vin_k(const float* __restrict__ V, const float* __restrict__ P,
                             float* __restrict__ VIN, int M) {
  int i = blockIdx.x * blockDim.x + threadIdx.x;
  if (i >= M * NDIM) return;
  int t = i >> 8, c = i & 255;
  VIN[i] = (c < DDIM) ? V[(size_t)t * DP + c] : P[(size_t)t * ENCS + (c - DDIM)];
}

__global__ void gate_apply_k(float* __restrict__ AO, const float* __restrict__ G, int M) {
  int i = blockIdx.x * blockDim.x + threadIdx.x;
  if (i >= M * NDIM) return;
  int t = i >> 8, h = (i & 255) >> 5;
  AO[i] *= G[t * NH + h];
}

__global__ void softmax4_k(float* __restrict__ GW, int M) {
  int r = blockIdx.x * blockDim.x + threadIdx.x;
  if (r >= M) return;
  float* g = GW + (size_t)r * NEXP;
  float mx = fmaxf(fmaxf(g[0], g[1]), fmaxf(g[2], g[3]));
  float e0 = __expf(g[0] - mx), e1 = __expf(g[1] - mx);
  float e2 = __expf(g[2] - mx), e3 = __expf(g[3] - mx);
  float inv = 1.0f / (e0 + e1 + e2 + e3);
  g[0] = e0 * inv; g[1] = e1 * inv; g[2] = e2 * inv; g[3] = e3 * inv;
}

__global__ void addip_k(float* __restrict__ A, const float* __restrict__ B, int n) {
  int i = blockIdx.x * blockDim.x + threadIdx.x;
  if (i < n) A[i] += B[i];
}

__global__ void copy_k(float* __restrict__ dst, const float* __restrict__ src, int n) {
  int i = blockIdx.x * blockDim.x + threadIdx.x;
  if (i < n) dst[i] = src[i];
}

__global__ void feat_asm_k(float* __restrict__ F, const float* __restrict__ O0,
                           const float* __restrict__ O1, const float* __restrict__ O2,
                           const float* __restrict__ O3, const float* __restrict__ P,
                           int M) {
  int i = blockIdx.x * blockDim.x + threadIdx.x;
  if (i >= M * FEATP) return;
  int t = i / FEATP, c = i - t * FEATP;
  float v = 0.0f;
  if (c < NBLK * DDIM) {
    int bi = c / DDIM, cc = c - bi * DDIM;
    const float* O = (bi == 0) ? O0 : (bi == 1) ? O1 : (bi == 2) ? O2 : O3;
    v = O[(size_t)t * DP + cc];
  } else if (c < FEATW) {
    v = P[(size_t)t * ENCS + (c - NBLK * DDIM)];
  }
  F[i] = v;
}

// ---------------- host orchestration ----------------
extern "C" void kernel_launch(void* const* d_in, const int* in_sizes, int n_in,
                              void* d_out, int out_size, void* d_ws, size_t ws_size,
                              hipStream_t stream) {
  (void)in_sizes; (void)n_in; (void)out_size; (void)ws_size;
  const float* state_in   = (const float*)d_in[0];
  const float* node_pos   = (const float*)d_in[1];
  const float* time_i     = (const float*)d_in[3];
  const float* conditions = (const float*)d_in[4];
  int pi = 5;
  const float* fv1_W1 = (const float*)d_in[pi++]; const float* fv1_b1 = (const float*)d_in[pi++];
  const float* fv1_W2 = (const float*)d_in[pi++]; const float* fv1_b2 = (const float*)d_in[pi++];
  const float* fvt_W1 = (const float*)d_in[pi++]; const float* fvt_b1 = (const float*)d_in[pi++];
  const float* fvt_W2 = (const float*)d_in[pi++]; const float* fvt_b2 = (const float*)d_in[pi++];
  const float* fvc_W1 = (const float*)d_in[pi++]; const float* fvc_b1 = (const float*)d_in[pi++];
  const float* fvc_W2 = (const float*)d_in[pi++]; const float* fvc_b2 = (const float*)d_in[pi++];
  const float* dec_W1 = (const float*)d_in[pi++]; const float* dec_b1 = (const float*)d_in[pi++];
  const float* dec_W2 = (const float*)d_in[pi++]; const float* dec_b2 = (const float*)d_in[pi++];
  const float* dec_W3 = (const float*)d_in[pi++]; const float* dec_b3 = (const float*)d_in[pi++];
  struct Blk {
    const float *ln1g,*ln1b,*Wq,*Wk,*Wv,*Wo,*bo,*Wg,*bg,*ln2g,*ln2b,
                *Wgate,*bgate,*We1,*be1,*We2,*be2,*Wpo,*bpo;
  } bp[NBLK];
  for (int b = 0; b < NBLK; ++b) {
    int base = 23 + b * 19;
    bp[b].ln1g  = (const float*)d_in[base + 0];
    bp[b].ln1b  = (const float*)d_in[base + 1];
    bp[b].Wq    = (const float*)d_in[base + 2];
    bp[b].Wk    = (const float*)d_in[base + 3];
    bp[b].Wv    = (const float*)d_in[base + 4];
    bp[b].Wo    = (const float*)d_in[base + 5];
    bp[b].bo    = (const float*)d_in[base + 6];
    bp[b].Wg    = (const float*)d_in[base + 7];
    bp[b].bg    = (const float*)d_in[base + 8];
    bp[b].ln2g  = (const float*)d_in[base + 9];
    bp[b].ln2b  = (const float*)d_in[base + 10];
    bp[b].Wgate = (const float*)d_in[base + 11];
    bp[b].bgate = (const float*)d_in[base + 12];
    bp[b].We1   = (const float*)d_in[base + 13];
    bp[b].be1   = (const float*)d_in[base + 14];
    bp[b].We2   = (const float*)d_in[base + 15];
    bp[b].be2   = (const float*)d_in[base + 16];
    bp[b].Wpo   = (const float*)d_in[base + 17];
    bp[b].bpo   = (const float*)d_in[base + 18];
  }

  const int M = MTOK;
  float* ws = (float*)d_ws;
  size_t off = 0;
  auto carve = [&](size_t n) { float* p = ws + off; off += (n + 31) & ~(size_t)31; return p; };

  float* POS   = carve((size_t)M * ENCS);
  float* TENC  = carve(32 * 32);
  float* CENC  = carve(32 * 352);
  float* CAT7  = carve((size_t)M * 32);
  float* TH    = carve(32 * DP);
  float* TV    = carve(32 * DP);
  float* CH    = carve(32 * DP);
  float* CV    = carve(32 * DP);
  float* V0    = carve((size_t)M * DP);
  float* OUTS[NBLK];
  for (int b = 0; b < NBLK; ++b) OUTS[b] = carve((size_t)M * DP);
  float* VIN   = carve((size_t)M * NDIM);
  float* XN    = carve((size_t)M * NDIM);
  float* Qb    = carve((size_t)M * NDIM);
  float* Kb    = carve((size_t)M * NDIM);
  float* Vb    = carve((size_t)M * NDIM);
  float* AO    = carve((size_t)M * NDIM);
  float* GATE  = carve((size_t)M * NH);
  float* GW    = carve((size_t)M * NEXP);
  float* HMOE  = carve((size_t)M * FFN);
  float* MOEACC= carve((size_t)M * NDIM);
  float* FEAT  = carve((size_t)M * FEATP);
  float* H1    = carve((size_t)M * DP);
  float* H2    = carve((size_t)M * DP);

  auto gemm = [&](const float* A, int lda, const float* W, const float* bias,
                  float* C, int ldc, int m, int k, int n, int act,
                  const float* rs, int rss, int rso, int accum, float osc) {
    int Kp = (k + 31) & ~31, Np = (n + 31) & ~31;
    _Float16* WP = (_Float16*)carve((size_t)Kp * Np / 2);
    int total = Kp * Np;
    pack_w<<<(total + 255) / 256, 256, 0, stream>>>(W, WP, k, n, Np, total);
    dim3 g((n + 31) / 32, (m + 31) / 32);
    gemm_wmma<<<g, 32, 0, stream>>>(A, lda, WP, Np, bias, C, ldc, m, k, n, act,
                                    rs, rss, rso, accum, osc);
  };
  auto ew = [&](int n) { return dim3((n + 255) / 256); };

  // ---- embeddings ----
  fourier_g<<<ew(M), 256, 0, stream>>>(node_pos, POS, M, SPC, ENCS);
  fourier_g<<<1, 32, 0, stream>>>(time_i, TENC, BSZ, 1, 32);
  fourier_g<<<1, 32, 0, stream>>>(conditions, CENC, BSZ, CDIM, 352);
  cat7_k<<<ew(M * 32), 256, 0, stream>>>(state_in, node_pos, CAT7, M);

  gemm(CAT7, 32, fv1_W1, fv1_b1, H1, DP, M, 7, DDIM, ACT_SILU, nullptr, 0, 0, 0, 1.0f);
  gemm(H1, DP, fv1_W2, fv1_b2, V0, DP, M, DDIM, DDIM, ACT_NONE, nullptr, 0, 0, 0, 1.0f);
  gemm(TENC, 32, fvt_W1, fvt_b1, TH, DP, BSZ, 11, DDIM, ACT_SILU, nullptr, 0, 0, 0, 1.0f);
  gemm(TH, DP, fvt_W2, fvt_b2, TV, DP, BSZ, DDIM, DDIM, ACT_NONE, nullptr, 0, 0, 0, 1.0f);
  gemm(CENC, 352, fvc_W1, fvc_b1, CH, DP, BSZ, 352, DDIM, ACT_SILU, nullptr, 0, 0, 0, 1.0f);
  gemm(CH, DP, fvc_W2, fvc_b2, CV, DP, BSZ, DDIM, DDIM, ACT_NONE, nullptr, 0, 0, 0, 1.0f);
  addbc_k<<<ew(M * DDIM), 256, 0, stream>>>(V0, TV, CV, M);

  // ---- transformer blocks ----
  const float* Vprev = V0;
  for (int b = 0; b < NBLK; ++b) {
    concat_vin_k<<<ew(M * NDIM), 256, 0, stream>>>(Vprev, POS, VIN, M);
    layernorm256<<<M / 8, 256, 0, stream>>>(VIN, bp[b].ln1g, bp[b].ln1b, XN, M);
    gemm(XN, NDIM, bp[b].Wq, nullptr, Qb, NDIM, M, NDIM, NDIM, ACT_NONE, nullptr, 0, 0, 0, 1.0f);
    gemm(XN, NDIM, bp[b].Wk, nullptr, Kb, NDIM, M, NDIM, NDIM, ACT_NONE, nullptr, 0, 0, 0, 1.0f);
    gemm(XN, NDIM, bp[b].Wv, nullptr, Vb, NDIM, M, NDIM, NDIM, ACT_NONE, nullptr, 0, 0, 0, 1.0f);
    gemm(XN, NDIM, bp[b].Wg, bp[b].bg, GATE, NH, M, NDIM, NH, ACT_SIGMOID, nullptr, 0, 0, 0, 1.0f);
    flash_attn<<<dim3(NTOK / 16, NH, BSZ), 32, 0, stream>>>(Qb, Kb, Vb, AO, NTOK);
    gate_apply_k<<<ew(M * NDIM), 256, 0, stream>>>(AO, GATE, M);
    gemm(AO, NDIM, bp[b].Wo, bp[b].bo, VIN, NDIM, M, NDIM, NDIM, ACT_NONE, nullptr, 0, 0, 1, 1.0f);
    layernorm256<<<M / 8, 256, 0, stream>>>(VIN, bp[b].ln2g, bp[b].ln2b, XN, M);
    gemm(XN, NDIM, bp[b].Wgate, bp[b].bgate, GW, NEXP, M, NDIM, NEXP, ACT_NONE, nullptr, 0, 0, 0, 1.0f);
    softmax4_k<<<ew(M), 256, 0, stream>>>(GW, M);
    for (int e = 0; e < NEXP; ++e) {
      gemm(XN, NDIM, bp[b].We1 + (size_t)e * NDIM * FFN, bp[b].be1 + (size_t)e * FFN,
           HMOE, FFN, M, NDIM, FFN, ACT_SILU, nullptr, 0, 0, 0, 1.0f);
      gemm(HMOE, FFN, bp[b].We2 + (size_t)e * FFN * NDIM, bp[b].be2 + (size_t)e * NDIM,
           MOEACC, NDIM, M, FFN, NDIM, ACT_NONE, GW, NEXP, e, (e > 0) ? 1 : 0, 1.0f);
    }
    addip_k<<<ew(M * NDIM), 256, 0, stream>>>(VIN, MOEACC, M * NDIM);
    copy_k<<<ew(M * DP), 256, 0, stream>>>(OUTS[b], Vprev, M * DP);
    gemm(VIN, NDIM, bp[b].Wpo, bp[b].bpo, OUTS[b], DP, M, NDIM, DDIM, ACT_NONE, nullptr, 0, 0, 1, 1.0f);
    Vprev = OUTS[b];
  }

  // ---- decoder ----
  feat_asm_k<<<ew(M * FEATP), 256, 0, stream>>>(FEAT, OUTS[0], OUTS[1], OUTS[2],
                                                OUTS[3], POS, M);
  gemm(FEAT, FEATP, dec_W1, dec_b1, H1, DP, M, FEATW, DDIM, ACT_SILU, nullptr, 0, 0, 0, 1.0f);
  gemm(H1, DP, dec_W2, dec_b2, H2, DP, M, DDIM, DDIM, ACT_SILU, nullptr, 0, 0, 0, 1.0f);
  copy_k<<<ew(M * OUTC), 256, 0, stream>>>((float*)d_out, state_in, M * OUTC);
  gemm(H2, DP, dec_W3, dec_b3, (float*)d_out, OUTC, M, DDIM, OUTC, ACT_NONE, nullptr, 0, 0, 1, DTs);
}
